// MultiHeadedAttention_39651138076920
// MI455X (gfx1250) — matrix operand order repaired and checked by reading, not compiled, not run on hardware
//
#include <hip/hip_runtime.h>

typedef __attribute__((ext_vector_type(16))) _Float16 v16h;
typedef __attribute__((ext_vector_type(8)))  _Float16 v8h;
typedef __attribute__((ext_vector_type(8)))  float    v8f;

// Problem constants (match reference)
constexpr int Bc  = 4;
constexpr int Sc  = 2048;
constexpr int Hc  = 1024;
constexpr int NHc = 16;
constexpr int DHc = 64;
constexpr int Mc  = Bc * Sc;  // 8192 rows for the projection GEMMs

// ---------------------------------------------------------------------------
// WMMA fragment helpers (f16, 16x16x32, wave32)
// A-matrix 16x32: lane holds row (lane&15); 16 halves cover K = base..base+7
// and 16+base..16+base+7 where base = 8*(lane>=16).   (ISA 7.12.2)
// B-matrix 32x16 (column-major source): lane holds column (lane&15); 16
// halves cover K = base..base+7 and 16+base..16+base+7 (same as A).
// C/D 16x16 f32: element r of v8f is row r + 8*(lane>=16), col lane&15.
// ---------------------------------------------------------------------------
static __device__ __forceinline__ v16h load_frag_a(const _Float16* p, int stride, int lane) {
  const _Float16* row = p + (size_t)(lane & 15) * stride;
  const int kb = (lane >> 4) << 3;  // 0 or 8
  union { v16h v; v8h h[2]; } u;
  u.h[0] = *(const v8h*)(row + kb);
  u.h[1] = *(const v8h*)(row + 16 + kb);
  return u.v;
}

static __device__ __forceinline__ v16h load_frag_b(const _Float16* p, int stride, int lane) {
  const _Float16* col = p + (size_t)(lane & 15) * stride;
  const int kb = (lane >> 4) << 3;  // 0 or 8
  union { v16h v; v8h h[2]; } u;
  u.h[0] = *(const v8h*)(col + kb);
  u.h[1] = *(const v8h*)(col + 16 + kb);
  return u.v;
}

static __device__ __forceinline__ v8f wmma_f16(v16h a, v16h b, v8f c) {
  return __builtin_amdgcn_wmma_f32_16x16x32_f16(false, a, false, b, (short)0, c,
                                                false, false);
}

// ---------------------------------------------------------------------------
// DPP16 row reductions (pure VALU, stay within 16-lane halves of the wave):
// xor1 = quad_perm(1,0,3,2)=0xB1, xor2 = quad_perm(2,3,0,1)=0x4E,
// cross-quad = row_half_mirror 0x141, cross-8 = row_mirror 0x140.
// ---------------------------------------------------------------------------
template <int CTRL>
static __device__ __forceinline__ float dppf(float x) {
  const int xi = __builtin_bit_cast(int, x);
  return __builtin_bit_cast(float,
      __builtin_amdgcn_update_dpp(xi, xi, CTRL, 0xF, 0xF, true));
}

static __device__ __forceinline__ float row16_max(float v) {
  v = fmaxf(v, dppf<0xB1>(v));
  v = fmaxf(v, dppf<0x4E>(v));
  v = fmaxf(v, dppf<0x141>(v));
  v = fmaxf(v, dppf<0x140>(v));
  return v;
}

static __device__ __forceinline__ float row16_sum(float v) {
  v += dppf<0xB1>(v);
  v += dppf<0x4E>(v);
  v += dppf<0x141>(v);
  v += dppf<0x140>(v);
  return v;
}

// ---------------------------------------------------------------------------
// Kernel: fp32 -> f16 cast (elementwise, float4 loads)
// ---------------------------------------------------------------------------
__global__ void cast_f32_to_f16(const float* __restrict__ in,
                                _Float16* __restrict__ out, int n) {
  int i = (blockIdx.x * blockDim.x + threadIdx.x) * 4;
  if (i + 3 < n) {
    float4 f = *(const float4*)(in + i);
    out[i + 0] = (_Float16)f.x;
    out[i + 1] = (_Float16)f.y;
    out[i + 2] = (_Float16)f.z;
    out[i + 3] = (_Float16)f.w;
  }
}

// Kernel: weight [K][N] fp32 -> transposed [N][K] f16 (so B-frags are contiguous)
__global__ void cast_transpose_w(const float* __restrict__ w,
                                 _Float16* __restrict__ wt) {
  int t = blockIdx.x * blockDim.x + threadIdx.x;  // 0 .. H*H-1
  int k = t >> 10;
  int n = t & (Hc - 1);
  wt[(size_t)n * Hc + k] = (_Float16)w[t];
}

// ---------------------------------------------------------------------------
// WMMA GEMM:  Y[M=8192, N=1024] = A_f16[M,1024] * W^T_f16 + bias
// Block = 256 threads = 8 waves; each wave -> 32 rows x 64 cols
// (2 A-frags x 4 B-frags -> 8 WMMAs per K step).
// MODE 0: store f16 head-split [BH][S][DH]   (Q, K)
// MODE 1: store f16 head-split transposed [BH][DH][S]   (V)
// MODE 2: store fp32 row-major [M][H]        (final output projection)
// ---------------------------------------------------------------------------
template <int MODE>
__global__ void gemm16(const _Float16* __restrict__ A,
                       const _Float16* __restrict__ WT,
                       const float* __restrict__ bias,
                       _Float16* __restrict__ outH,
                       float* __restrict__ outF) {
  const int lane = threadIdx.x & 31;
  const int wave = threadIdx.x >> 5;               // 0..7
  const int m0 = blockIdx.y * 256 + wave * 32;     // row tile (32 rows)
  const int n0 = blockIdx.x * 64;                  // col tile

  const v8f zero = {0.f, 0.f, 0.f, 0.f, 0.f, 0.f, 0.f, 0.f};
  v8f acc[2][4] = {{zero, zero, zero, zero}, {zero, zero, zero, zero}};

  const _Float16* arow0 = A + (size_t)m0 * Hc;
  const _Float16* arow1 = A + (size_t)(m0 + 16) * Hc;
  for (int k0 = 0; k0 < Hc; k0 += 32) {
    v16h a0 = load_frag_a(arow0 + k0, Hc, lane);
    v16h a1 = load_frag_a(arow1 + k0, Hc, lane);
#pragma unroll
    for (int t = 0; t < 4; ++t) {
      v16h b = load_frag_b(WT + (size_t)(n0 + 16 * t) * Hc + k0, Hc, lane);
      acc[0][t] = wmma_f16(a0, b, acc[0][t]);
      acc[1][t] = wmma_f16(a1, b, acc[1][t]);
    }
  }

  const int colbase = lane & 15;
  const int rowoff  = (lane >> 4) << 3;  // 0 or 8
#pragma unroll
  for (int i = 0; i < 2; ++i) {
#pragma unroll
    for (int t = 0; t < 4; ++t) {
      const int n = n0 + 16 * t + colbase;
      const float bv = bias[n];
#pragma unroll
      for (int r = 0; r < 8; ++r) {
        const int m = m0 + 16 * i + r + rowoff;
        const float val = acc[i][t][r] + bv;
        if (MODE == 2) {
          outF[(size_t)m * Hc + n] = val;
        } else {
          const int b = m >> 11, s = m & (Sc - 1);
          const int h = n >> 6,  d = n & (DHc - 1);
          if (MODE == 0)
            outH[(((size_t)(b * NHc + h)) * Sc + s) * DHc + d] = (_Float16)val;
          else
            outH[(((size_t)(b * NHc + h)) * DHc + d) * Sc + s] = (_Float16)val;
        }
      }
    }
  }
}

// ---------------------------------------------------------------------------
// Flash attention: one (b,h) per blockIdx.x, 64 queries per block (16/wave).
// 64 keys per iteration: 8 score WMMAs, online softmax in the exp2 domain
// with DPP row reductions, P->LDS->two A-frags, 8 PV WMMAs.
// Q/K are [BH][S][64] f16, V is transposed [BH][64][S] f16.
// ---------------------------------------------------------------------------
__global__ void flash_attn(const _Float16* __restrict__ Q,
                           const _Float16* __restrict__ Km,
                           const _Float16* __restrict__ VT,
                           _Float16* __restrict__ O) {
  const int lane = threadIdx.x & 31;
  const int wave = threadIdx.x >> 5;           // 0..3
  const int bh   = blockIdx.x;                 // 0..63
  const int q0   = blockIdx.y * 64 + wave * 16;

  __shared__ _Float16 pbuf[4][16][64];         // per-wave 16x64 P tile (8 KB)

  const _Float16* qbase = Q + ((size_t)bh * Sc + q0) * DHc;
  const v16h qf0 = load_frag_a(qbase, DHc, lane);
  const v16h qf1 = load_frag_a(qbase + 32, DHc, lane);

  const v8f zero = {0.f, 0.f, 0.f, 0.f, 0.f, 0.f, 0.f, 0.f};
  v8f oacc[4] = {zero, zero, zero, zero};
  float mrow[8], lrow[8];
#pragma unroll
  for (int r = 0; r < 8; ++r) { mrow[r] = -1e30f; lrow[r] = 0.f; }

  // scores kept in exp2 domain: scale = (1/sqrt(64)) * log2(e)
  const float qk_scale = 0.125f * 1.44269504088896f;
  const int colbase = lane & 15;
  const int rowoff  = (lane >> 4) << 3;

  for (int k0 = 0; k0 < Sc; k0 += 64) {
    const _Float16* kbase = Km + ((size_t)bh * Sc + k0) * DHc;
    if (k0 + 64 < Sc) __builtin_prefetch(kbase + 64 * DHc, 0, 1);

    // scores: four 16x16 tiles covering keys k0 .. k0+63, K-dim = DH = 64
    v8f sc[4];
#pragma unroll
    for (int t = 0; t < 4; ++t) {
      const _Float16* kb = kbase + 16 * t * DHc;
      v8f s = zero;
      s = wmma_f16(qf0, load_frag_b(kb, DHc, lane), s);
      s = wmma_f16(qf1, load_frag_b(kb + 32, DHc, lane), s);
      sc[t] = s;
    }

    // online softmax per output row (DPP reductions across 16-lane halves)
#pragma unroll
    for (int r = 0; r < 8; ++r) {
      float s0 = sc[0][r] * qk_scale;
      float s1 = sc[1][r] * qk_scale;
      float s2 = sc[2][r] * qk_scale;
      float s3 = sc[3][r] * qk_scale;
      const float rm = row16_max(fmaxf(fmaxf(s0, s1), fmaxf(s2, s3)));
      const float mnew = fmaxf(mrow[r], rm);
      const float p0 = __builtin_amdgcn_exp2f(s0 - mnew);
      const float p1 = __builtin_amdgcn_exp2f(s1 - mnew);
      const float p2 = __builtin_amdgcn_exp2f(s2 - mnew);
      const float p3 = __builtin_amdgcn_exp2f(s3 - mnew);
      const float rs = row16_sum((p0 + p1) + (p2 + p3));
      const float alpha = __builtin_amdgcn_exp2f(mrow[r] - mnew);
      lrow[r] = lrow[r] * alpha + rs;
      mrow[r] = mnew;
#pragma unroll
      for (int t = 0; t < 4; ++t) oacc[t][r] *= alpha;
      const int prow = r + rowoff;
      pbuf[wave][prow][colbase]      = (_Float16)p0;
      pbuf[wave][prow][16 + colbase] = (_Float16)p1;
      pbuf[wave][prow][32 + colbase] = (_Float16)p2;
      pbuf[wave][prow][48 + colbase] = (_Float16)p3;
    }

    // P (16x64 f16) from LDS as two A-fragments, then O += P * V
    const v16h pf0 = load_frag_a(&pbuf[wave][0][0], 64, lane);
    const v16h pf1 = load_frag_a(&pbuf[wave][0][0] + 32, 64, lane);
#pragma unroll
    for (int t = 0; t < 4; ++t) {
      const _Float16* vb = VT + ((size_t)bh * DHc + 16 * t) * Sc + k0;
      oacc[t] = wmma_f16(pf0, load_frag_b(vb, Sc, lane), oacc[t]);
      oacc[t] = wmma_f16(pf1, load_frag_b(vb + 32, Sc, lane), oacc[t]);
    }
  }

  // normalize and store to [B][S][H] f16 for the output projection
  const int b = bh >> 4, h = bh & 15;
#pragma unroll
  for (int r = 0; r < 8; ++r) {
    const float inv = 1.0f / lrow[r];
    const int s = q0 + r + rowoff;
#pragma unroll
    for (int t = 0; t < 4; ++t) {
      const int d = 16 * t + colbase;
      O[((size_t)(b * Sc + s)) * Hc + h * DHc + d] = (_Float16)(oacc[t][r] * inv);
    }
  }
}

// ---------------------------------------------------------------------------
// Launcher
// ---------------------------------------------------------------------------
extern "C" void kernel_launch(void* const* d_in, const int* in_sizes, int n_in,
                              void* d_out, int out_size, void* d_ws, size_t ws_size,
                              hipStream_t stream) {
  (void)in_sizes; (void)n_in; (void)out_size; (void)ws_size;
  const float* key   = (const float*)d_in[0];
  const float* value = (const float*)d_in[1];
  const float* query = (const float*)d_in[2];
  const float* wq = (const float*)d_in[3];
  const float* bq = (const float*)d_in[4];
  const float* wk = (const float*)d_in[5];
  const float* bk = (const float*)d_in[6];
  const float* wv = (const float*)d_in[7];
  const float* bv = (const float*)d_in[8];
  const float* wo = (const float*)d_in[9];
  const float* bo = (const float*)d_in[10];

  // Workspace carve-up (total 120 MB of f16 scratch)
  char* ws = (char*)d_ws;
  const size_t WBYTES = (size_t)Hc * Hc * 2;   // 2 MB per weight
  const size_t XBYTES = (size_t)Mc * Hc * 2;   // 16 MB per activation tensor
  _Float16* wqT = (_Float16*)(ws + 0 * WBYTES);
  _Float16* wkT = (_Float16*)(ws + 1 * WBYTES);
  _Float16* wvT = (_Float16*)(ws + 2 * WBYTES);
  _Float16* woT = (_Float16*)(ws + 3 * WBYTES);
  char* pa = ws + 4 * WBYTES;
  _Float16* xq  = (_Float16*)(pa + 0 * XBYTES);
  _Float16* xk  = (_Float16*)(pa + 1 * XBYTES);
  _Float16* xv  = (_Float16*)(pa + 2 * XBYTES);
  _Float16* Qh  = (_Float16*)(pa + 3 * XBYTES);
  _Float16* Kh  = (_Float16*)(pa + 4 * XBYTES);
  _Float16* VTh = (_Float16*)(pa + 5 * XBYTES);
  _Float16* Oh  = (_Float16*)(pa + 6 * XBYTES);

  const int nX = Mc * Hc;         // 8,388,608
  const int nW = Hc * Hc;         // 1,048,576
  cast_f32_to_f16<<<dim3(nX / 4 / 256), dim3(256), 0, stream>>>(query, xq, nX);
  cast_f32_to_f16<<<dim3(nX / 4 / 256), dim3(256), 0, stream>>>(key,   xk, nX);
  cast_f32_to_f16<<<dim3(nX / 4 / 256), dim3(256), 0, stream>>>(value, xv, nX);
  cast_transpose_w<<<dim3(nW / 256), dim3(256), 0, stream>>>(wq, wqT);
  cast_transpose_w<<<dim3(nW / 256), dim3(256), 0, stream>>>(wk, wkT);
  cast_transpose_w<<<dim3(nW / 256), dim3(256), 0, stream>>>(wv, wvT);
  cast_transpose_w<<<dim3(nW / 256), dim3(256), 0, stream>>>(wo, woT);

  const dim3 gg(Hc / 64, Mc / 256);  // (16, 32)
  gemm16<0><<<gg, dim3(256), 0, stream>>>(xq, wqT, bq, Qh,  nullptr);
  gemm16<0><<<gg, dim3(256), 0, stream>>>(xk, wkT, bk, Kh,  nullptr);
  gemm16<1><<<gg, dim3(256), 0, stream>>>(xv, wvT, bv, VTh, nullptr);

  flash_attn<<<dim3(Bc * NHc, Sc / 64), dim3(128), 0, stream>>>(Qh, Kh, VTh, Oh);

  gemm16<2><<<gg, dim3(256), 0, stream>>>(Oh, woT, bo, nullptr, (float*)d_out);
}
